// GPT2Attention_56573309223275
// MI455X (gfx1250) — matrix operand order repaired and checked by reading, not compiled, not run on hardware
//
#include <hip/hip_runtime.h>
#include <hip/hip_bf16.h>

// ---------------------------------------------------------------------------
// GPT-2 attention block for MI455X (gfx1250), wave32.
//   - v_wmma_f32_16x16x32_bf16 for all matmuls
//   - global_load_async_to_lds_b128 (ASYNCcnt) double-buffered A-tile staging
// ---------------------------------------------------------------------------

typedef __bf16 bf16_t;
typedef __attribute__((ext_vector_type(16))) __bf16 v16bf;
typedef __attribute__((ext_vector_type(8)))  __bf16 v8bf;
typedef __attribute__((ext_vector_type(8)))  float  v8f;

union AFrag { v16bf v; v8bf h[2]; };

#define S_LEN 2048
#define EMB   1024
#define NHEAD 16
#define HDIM  64
#define NB    2
#define BSROWS (NB * S_LEN)   // 4096 rows total

#define BLK_M 64              // block M tile (shared A staged in LDS)
#define BLK_N 128             // 4 waves * 32 cols each

// ---------------------------------------------------------------------------
// Conversion / transpose kernels
// ---------------------------------------------------------------------------
__global__ void cvt_x_bf16(const float* __restrict__ x, bf16_t* __restrict__ xbf, int n) {
    int i = blockIdx.x * blockDim.x + threadIdx.x;
    if (i < n) xbf[i] = (bf16_t)x[i];
}

__global__ void tr_qkvw_bf16(const float* __restrict__ w, bf16_t* __restrict__ wT) {
    int i = blockIdx.x * blockDim.x + threadIdx.x;
    if (i < EMB * 3 * EMB) {
        int k = i / (3 * EMB);
        int n = i - k * (3 * EMB);
        wT[(size_t)n * EMB + k] = (bf16_t)w[i];
    }
}

__global__ void tr_projw_bf16(const float* __restrict__ w, bf16_t* __restrict__ pT) {
    int i = blockIdx.x * blockDim.x + threadIdx.x;
    if (i < EMB * EMB) {
        int k = i >> 10;
        int n = i & 1023;
        pT[(size_t)n * EMB + k] = (bf16_t)w[i];
    }
}

// ---------------------------------------------------------------------------
// WMMA fragment helpers (C layout per CDNA5 ISA sec 7.12.2; A/B use this code's matched contiguous k)
//   A (16x32 bf16): lane row m = lane&15; elem e -> k = 16*half + e (contig)
//   B (32x16 bf16): lane col n = lane&15; elem e -> k = 16*half + e (contig)
//   C (16x16 f32) : VGPR r -> row m = 8*half + r; col n = lane&15
// ---------------------------------------------------------------------------
__device__ __forceinline__ AFrag load_a_frag(const bf16_t* row, int half) {
    AFrag a;
    a.h[0] = *(const v8bf*)(row + 16 * half);
    a.h[1] = *(const v8bf*)(row + 16 * half + 8);
    return a;
}

// Issue one async 16B global->LDS chunk copy (per-lane addresses, ASYNCcnt++).
__device__ __forceinline__ void async_copy16(const bf16_t* gp, const bf16_t* lp) {
    unsigned int       la = (unsigned int)(unsigned long long)lp;  // flat->LDS: addr[31:0]
    unsigned long long ga = (unsigned long long)gp;
    asm volatile("global_load_async_to_lds_b128 %0, %1, off"
                 :: "v"(la), "v"(ga) : "memory");
}

// ---------------------------------------------------------------------------
// Tiled GEMM mainloop: block = 4 waves, tile BLK_M x BLK_N, K-step 32.
// A tile (64x32 bf16 = 4KB) async-staged into LDS, double buffered, shared by
// all 4 waves.  B rows (transposed weights) read directly (L2-resident).
// acc[ms][t]: wave tile 64(M) x 32(N): ms = M sub-tile, t = N sub-tile.
// ---------------------------------------------------------------------------
#define GEMM_MAINLOOP(Aptr, BTptr, Kdim)                                          \
    __shared__ bf16_t tileA[2][BLK_M * 32];                                       \
    const int tid  = threadIdx.x;                                                 \
    const int wv   = tid >> 5;                                                    \
    const int lane = tid & 31;                                                    \
    const int half = lane >> 4, lm = lane & 15;                                   \
    const int m0  = blockIdx.y * BLK_M;                                           \
    const int nw0 = blockIdx.x * BLK_N + wv * 32;                                 \
    v8f acc[4][2] = {};                                                           \
    {   /* prologue: stage 0 */                                                   \
        _Pragma("unroll")                                                         \
        for (int u = 0; u < 2; ++u) {                                             \
            int c = tid * 2 + u;                                                  \
            int row = c >> 2, col8 = (c & 3) * 8;                                 \
            async_copy16(Aptr + (size_t)(m0 + row) * Kdim + col8,                 \
                         &tileA[0][row * 32 + col8]);                             \
        }                                                                         \
    }                                                                             \
    for (int kk = 0; kk < Kdim; kk += 32) {                                       \
        const int buf = (kk >> 5) & 1;                                            \
        if (kk + 32 < Kdim) {                                                     \
            _Pragma("unroll")                                                     \
            for (int u = 0; u < 2; ++u) {                                         \
                int c = tid * 2 + u;                                              \
                int row = c >> 2, col8 = (c & 3) * 8;                             \
                async_copy16(Aptr + (size_t)(m0 + row) * Kdim + kk + 32 + col8,   \
                             &tileA[buf ^ 1][row * 32 + col8]);                   \
            }                                                                     \
            asm volatile("s_wait_asynccnt 0x2" ::: "memory");                     \
        } else {                                                                  \
            asm volatile("s_wait_asynccnt 0x0" ::: "memory");                     \
        }                                                                         \
        __syncthreads();                                                          \
        v16bf bfrag[2];                                                           \
        _Pragma("unroll")                                                         \
        for (int t = 0; t < 2; ++t)                                               \
            bfrag[t] = *(const v16bf*)(BTptr + (size_t)(nw0 + t * 16 + lm) * Kdim \
                                       + kk + 16 * half);                         \
        _Pragma("unroll")                                                         \
        for (int ms = 0; ms < 4; ++ms) {                                          \
            AFrag a;                                                              \
            const bf16_t* lr = &tileA[buf][(ms * 16 + lm) * 32];                  \
            a.h[0] = *(const v8bf*)(lr + 16 * half);                              \
            a.h[1] = *(const v8bf*)(lr + 16 * half + 8);                          \
            _Pragma("unroll")                                                     \
            for (int t = 0; t < 2; ++t)                                           \
                acc[ms][t] = __builtin_amdgcn_wmma_f32_16x16x32_bf16(             \
                    false, a.v, false, bfrag[t], (short)0, acc[ms][t],            \
                    false, false);                                                \
        }                                                                         \
        __syncthreads();                                                          \
    }

// ---------------------------------------------------------------------------
// QKV GEMM:  qkv = x @ qkv_w + qkv_b      (M=4096, N=3072, K=1024)
// Scatters: q,k -> bf16 [B,H,S,D];  v -> bf16 vT [B,H,D,S];  k,v f32 -> d_out
// ---------------------------------------------------------------------------
__global__ __launch_bounds__(128) void qkv_gemm(
    const bf16_t* __restrict__ xbf, const bf16_t* __restrict__ wT,
    const float* __restrict__ qkv_b,
    bf16_t* __restrict__ qbf, bf16_t* __restrict__ kbf, bf16_t* __restrict__ vT,
    float* __restrict__ outk, float* __restrict__ outv)
{
    GEMM_MAINLOOP(xbf, wT, EMB)

#pragma unroll
    for (int ms = 0; ms < 4; ++ms) {
#pragma unroll
        for (int t = 0; t < 2; ++t) {
            const int n    = nw0 + t * 16 + lm;
            const float bs = qkv_b[n];
            const int part = n >> 10;          // 0=q 1=k 2=v
            const int rem  = n & 1023;
            const int hh   = rem >> 6;
            const int dd   = rem & 63;
#pragma unroll
            for (int r = 0; r < 8; ++r) {
                const int grow = m0 + ms * 16 + 8 * half + r;  // row in [0,4096)
                const int bb   = grow >> 11;
                const int ss   = grow & 2047;
                const float val = acc[ms][t][r] + bs;
                const size_t hs = (size_t)(bb * NHEAD + hh) * S_LEN + ss;
                if (part == 0) {
                    qbf[hs * HDIM + dd] = (bf16_t)val;
                } else if (part == 1) {
                    kbf[hs * HDIM + dd]  = (bf16_t)val;
                    outk[hs * HDIM + dd] = val;
                } else {
                    vT[((size_t)(bb * NHEAD + hh) * HDIM + dd) * S_LEN + ss] = (bf16_t)val;
                    outv[hs * HDIM + dd] = val;
                }
            }
        }
    }
}

// ---------------------------------------------------------------------------
// Attention: one wave per (b, h, 16-row q tile).
// Scores (WMMA) -> 128KB LDS -> causal softmax -> bf16 probs in place -> PV (WMMA).
// ---------------------------------------------------------------------------
__global__ __launch_bounds__(32) void attn_kernel(
    const bf16_t* __restrict__ qbf, const bf16_t* __restrict__ kbf,
    const bf16_t* __restrict__ vT, bf16_t* __restrict__ attn_out)
{
    __shared__ float sc[16 * S_LEN];   // 128 KB (320 KB/WGP available on CDNA5)

    const int lane = threadIdx.x;
    const int half = lane >> 4, lm = lane & 15;

    const int bid = blockIdx.x;                  // B*H*(S/16) = 4096 blocks
    const int qt  = bid & 127;
    const int hh  = (bid >> 7) & 15;
    const int bb  = bid >> 11;
    const int q0  = qt * 16;
    const int kvend = (q0 + 16 + 31) & ~31;      // padded causal extent

    const bf16_t* qbase = qbf + (size_t)(bb * NHEAD + hh) * S_LEN * HDIM;
    const bf16_t* kbase = kbf + (size_t)(bb * NHEAD + hh) * S_LEN * HDIM;
    const bf16_t* vbase = vT  + (size_t)(bb * NHEAD + hh) * HDIM * S_LEN;

    // ---- scores: S[16][0..q0+15] = (Q Kt) * 1/sqrt(64), causal-masked ----
    const bf16_t* arow = qbase + (size_t)(q0 + lm) * HDIM;
    for (int n0 = 0; n0 <= q0; n0 += 16) {
        v8f c = {};
#pragma unroll
        for (int kk = 0; kk < HDIM; kk += 32) {
            AFrag a = load_a_frag(arow + kk, half);
            v16bf b = *(const v16bf*)(kbase + (size_t)(n0 + lm) * HDIM + kk + 16 * half);
            c = __builtin_amdgcn_wmma_f32_16x16x32_bf16(
                false, a.v, false, b, (short)0, c, false, false);
        }
#pragma unroll
        for (int r = 0; r < 8; ++r) {
            const int m   = 8 * half + r;
            const int col = n0 + lm;
            sc[m * S_LEN + col] = (col <= q0 + m) ? c[r] * 0.125f : -3.0e38f;
        }
    }
    __syncthreads();

    // ---- softmax per row (lanes 0..15), convert to bf16 probs in place ----
    if (lane < 16) {
        const int i = lane;
        const int bound = q0 + i;                 // inclusive causal bound
        float* row = sc + i * S_LEN;
        float mx = -3.0e38f;
        for (int j = 0; j <= bound; ++j) mx = fmaxf(mx, row[j]);
        float sum = 0.f;
        for (int j = 0; j <= bound; ++j) { float e = __expf(row[j] - mx); sum += e; row[j] = e; }
        const float inv = 1.f / sum;
        bf16_t* prow = (bf16_t*)row;              // write @2j < read @4j: safe in place
        for (int j = 0; j <= bound; ++j) prow[j] = (bf16_t)(row[j] * inv);
        for (int j = bound + 1; j < kvend; ++j) prow[j] = (bf16_t)0.f;
    }
    __syncthreads();

    // ---- PV: out[16][64] = P @ V,  A from LDS (ds_load_b128), B = vT rows ----
    v8f acc[4] = {};
    const bf16_t* prow = (const bf16_t*)(sc + (size_t)lm * S_LEN);
    for (int j0 = 0; j0 < kvend; j0 += 32) {
        AFrag a;
        a.h[0] = *(const v8bf*)(prow + j0 + 16 * half);
        a.h[1] = *(const v8bf*)(prow + j0 + 16 * half + 8);
#pragma unroll
        for (int t = 0; t < 4; ++t) {
            v16bf b = *(const v16bf*)(vbase + (size_t)(t * 16 + lm) * S_LEN + j0 + 16 * half);
            acc[t] = __builtin_amdgcn_wmma_f32_16x16x32_bf16(
                false, a.v, false, b, (short)0, acc[t], false, false);
        }
    }
#pragma unroll
    for (int t = 0; t < 4; ++t) {
#pragma unroll
        for (int r = 0; r < 8; ++r) {
            const int srow = q0 + 8 * half + r;
            const int d    = t * 16 + lm;
            attn_out[(size_t)(bb * S_LEN + srow) * EMB + hh * HDIM + d] = (bf16_t)acc[t][r];
        }
    }
}

// ---------------------------------------------------------------------------
// Projection GEMM: out = attn_out @ proj_w + proj_b   (M=4096, N=1024, K=1024)
// ---------------------------------------------------------------------------
__global__ __launch_bounds__(128) void proj_gemm(
    const bf16_t* __restrict__ att, const bf16_t* __restrict__ pT,
    const float* __restrict__ proj_b, float* __restrict__ out)
{
    GEMM_MAINLOOP(att, pT, EMB)

#pragma unroll
    for (int ms = 0; ms < 4; ++ms) {
#pragma unroll
        for (int t = 0; t < 2; ++t) {
            const int n    = nw0 + t * 16 + lm;
            const float bs = proj_b[n];
#pragma unroll
            for (int r = 0; r < 8; ++r) {
                const int grow = m0 + ms * 16 + 8 * half + r;
                out[(size_t)grow * EMB + n] = acc[ms][t][r] + bs;
            }
        }
    }
}

// ---------------------------------------------------------------------------
// Launch
// ---------------------------------------------------------------------------
extern "C" void kernel_launch(void* const* d_in, const int* in_sizes, int n_in,
                              void* d_out, int out_size, void* d_ws, size_t ws_size,
                              hipStream_t stream) {
    const float* x      = (const float*)d_in[0];   // [2,2048,1024]
    const float* qkv_w  = (const float*)d_in[1];   // [1024,3072]
    const float* qkv_b  = (const float*)d_in[2];   // [3072]
    const float* proj_w = (const float*)d_in[3];   // [1024,1024]
    const float* proj_b = (const float*)d_in[4];   // [1024]

    float* out  = (float*)d_out;                   // [2,2048,1024]
    float* outk = out  + (size_t)BSROWS * EMB;     // [2,16,2048,64]
    float* outv = outk + (size_t)BSROWS * EMB;     // [2,16,2048,64]

    // workspace layout (bytes)
    char* ws = (char*)d_ws;
    bf16_t* xbf   = (bf16_t*)(ws);                                 //  8 MB
    bf16_t* qkvwT = (bf16_t*)(ws + (8u  << 20));                   //  6 MB
    bf16_t* projT = (bf16_t*)(ws + (14u << 20));                   //  2 MB
    bf16_t* qbf   = (bf16_t*)(ws + (16u << 20));                   //  8 MB
    bf16_t* kbf   = (bf16_t*)(ws + (24u << 20));                   //  8 MB
    bf16_t* vT    = (bf16_t*)(ws + (32u << 20));                   //  8 MB
    bf16_t* att   = (bf16_t*)(ws + (40u << 20));                   //  8 MB  (48 MB total)

    // 1) precision conversion / weight transposition
    cvt_x_bf16  <<<(BSROWS * EMB + 255) / 256, 256, 0, stream>>>(x, xbf, BSROWS * EMB);
    tr_qkvw_bf16<<<(EMB * 3 * EMB + 255) / 256, 256, 0, stream>>>(qkv_w, qkvwT);
    tr_projw_bf16<<<(EMB * EMB + 255) / 256, 256, 0, stream>>>(proj_w, projT);

    // 2) qkv GEMM: block tile 64x128 (4 waves), grid = (3072/128, 4096/64)
    qkv_gemm<<<dim3(24, 64), 128, 0, stream>>>(xbf, qkvwT, qkv_b,
                                               qbf, kbf, vT, outk, outv);

    // 3) attention: one wave per (b, h, q-tile)
    attn_kernel<<<NB * NHEAD * (S_LEN / 16), 32, 0, stream>>>(qbf, kbf, vT, att);

    // 4) output projection: block tile 64x128, grid = (1024/128, 4096/64)
    proj_gemm<<<dim3(8, 64), 128, 0, stream>>>(att, projT, proj_b, out);
}
